// SparseAttention_11587821765351
// MI455X (gfx1250) — matrix operand order repaired and checked by reading, not compiled, not run on hardware
//
#include <hip/hip_runtime.h>
#include <hip/hip_bf16.h>
#include <math.h>

// ---------------------------------------------------------------------------
// Problem constants (from reference): B=4, L=4096, Hd=512, H=8, D=64, S=u=45
// ---------------------------------------------------------------------------
#define BB    4
#define LL    4096
#define HD    512
#define NH    8
#define DD    64
#define SS    45
#define UP    48            // 45 padded to 3 WMMA row tiles
#define MTOT  (BB * LL)     // 16384 rows for the dense GEMMs
#define BH    (BB * NH)     // 32

typedef __bf16 bf16_t;
typedef bf16_t v16bf __attribute__((ext_vector_type(16)));
typedef float  v8f   __attribute__((ext_vector_type(8)));

union FragBF16 { v16bf v; uint4 q[2]; };

__device__ __forceinline__ v8f wmma_bf16(v16bf a, v16bf b, v8f c) {
    // v_wmma_f32_16x16x32_bf16: D = A(16x32) * B(32x16) + C(16x16 f32)
    return __builtin_amdgcn_wmma_f32_16x16x32_bf16(
        false, a, false, b, (short)0, c, false, false);
}

// CDNA5 async global->LDS copy (16B per lane), tracked by ASYNCcnt.
__device__ __forceinline__ void async_copy_b128(const void* gptr, void* lds_ptr) {
    uint32_t lds_off = (uint32_t)(uintptr_t)lds_ptr;          // LDS aperture: [31:0] = offset
    unsigned long long ga = (unsigned long long)(uintptr_t)gptr;
    asm volatile("global_load_async_to_lds_b128 %0, %1, off"
                 :: "v"(lds_off), "v"(ga) : "memory");
}
__device__ __forceinline__ void wait_async0() {
    asm volatile("s_wait_asynccnt 0x0" ::: "memory");
}

// ---------------------------------------------------------------------------
// 1) f32 -> bf16 conversions
// ---------------------------------------------------------------------------
__global__ __launch_bounds__(256) void convert_x_kernel(const float* __restrict__ x,
                                                        bf16_t* __restrict__ xb) {
    size_t i = (size_t)blockIdx.x * 256 + threadIdx.x;   // grid covers MTOT*HD
    xb[i] = (bf16_t)x[i];
}

// W (k-major [k][n]) -> WT bf16 ([n][k]) for all four weights (blockIdx.y)
__global__ __launch_bounds__(256) void convert_w_kernel(const float* __restrict__ Wq,
                                                        const float* __restrict__ Wk,
                                                        const float* __restrict__ Wv,
                                                        const float* __restrict__ Wo,
                                                        bf16_t* __restrict__ WT) {
    const float* W = (blockIdx.y == 0) ? Wq : (blockIdx.y == 1) ? Wk
                   : (blockIdx.y == 2) ? Wv : Wo;
    int i = blockIdx.x * 256 + threadIdx.x;              // 0 .. 512*512-1
    int k = i >> 9, n = i & 511;
    WT[(size_t)blockIdx.y * HD * HD + (size_t)n * HD + k] = (bf16_t)W[i];
}

// ---------------------------------------------------------------------------
// 2) Tiled bf16 WMMA GEMM: C(MTOT x 512) = A_bf16 @ W  (W given as WT [n][k])
//    Double-buffered LDS, async global->LDS staging.
//    mode 0: plain f32 row-major output (final projection)
//    mode 1: blockIdx.z selects Wq/Wk/Wv; output scattered to (B,H,L,D) f32,
//            with a bf16 mirror for k.
// ---------------------------------------------------------------------------
#define LDT 40   // LDS row stride (bf16 elems); 80B rows keep 16B alignment

__global__ __launch_bounds__(256) void gemm_bf16_kernel(
    const bf16_t* __restrict__ A,
    const bf16_t* __restrict__ WT0, const bf16_t* __restrict__ WT1,
    const bf16_t* __restrict__ WT2,
    float* __restrict__ dst0, float* __restrict__ dst1, float* __restrict__ dst2,
    bf16_t* __restrict__ kb, float* __restrict__ out_plain, int mode)
{
    __shared__ bf16_t sA[2][128 * LDT];
    __shared__ bf16_t sB[2][128 * LDT];

    const int tid  = threadIdx.x;
    const int lane = tid & 31;
    const int wid  = tid >> 5;          // 8 waves
    const int wm   = wid >> 2;          // 0..1 : 64-row slice
    const int wn   = wid & 3;           // 0..3 : 32-col slice
    const int m0   = blockIdx.x * 128;
    const int n0   = blockIdx.y * 128;
    const int which = blockIdx.z;

    const bf16_t* WT = (mode == 0) ? WT0
                     : (which == 0) ? WT0 : (which == 1) ? WT1 : WT2;

    // per-thread staging chunk coordinates (2 chunks of A, 2 of B per tile)
    const int r0 = tid >> 2,           c0 = tid & 3;          // chunk tid
    const int r1 = (tid + 256) >> 2,   c1 = (tid + 256) & 3;  // chunk tid+256

    v8f acc[4][2] = {};

    // prologue: stage K-tile 0 into buffer 0
    {
        async_copy_b128(A  + (size_t)(m0 + r0) * HD + c0 * 8, &sA[0][r0 * LDT + c0 * 8]);
        async_copy_b128(A  + (size_t)(m0 + r1) * HD + c1 * 8, &sA[0][r1 * LDT + c1 * 8]);
        async_copy_b128(WT + (size_t)(n0 + r0) * HD + c0 * 8, &sB[0][r0 * LDT + c0 * 8]);
        async_copy_b128(WT + (size_t)(n0 + r1) * HD + c1 * 8, &sB[0][r1 * LDT + c1 * 8]);
    }
    wait_async0();
    __syncthreads();

    const int rsel = lane & 15;
    const int hsel = lane >> 4;

    for (int kt = 0; kt < HD / 32; ++kt) {
        const int cur = kt & 1;
        if (kt + 1 < HD / 32) {   // prefetch next tile into the other buffer
            const int k0 = (kt + 1) * 32;
            async_copy_b128(A  + (size_t)(m0 + r0) * HD + k0 + c0 * 8, &sA[cur ^ 1][r0 * LDT + c0 * 8]);
            async_copy_b128(A  + (size_t)(m0 + r1) * HD + k0 + c1 * 8, &sA[cur ^ 1][r1 * LDT + c1 * 8]);
            async_copy_b128(WT + (size_t)(n0 + r0) * HD + k0 + c0 * 8, &sB[cur ^ 1][r0 * LDT + c0 * 8]);
            async_copy_b128(WT + (size_t)(n0 + r1) * HD + k0 + c1 * 8, &sB[cur ^ 1][r1 * LDT + c1 * 8]);
        }

        FragBF16 afrag[4], bfrag[2];
        #pragma unroll
        for (int mi = 0; mi < 4; ++mi) {
            const bf16_t* p = &sA[cur][(wm * 64 + mi * 16 + rsel) * LDT + hsel * 8];
            afrag[mi].q[0] = *(const uint4*)p;
            afrag[mi].q[1] = *(const uint4*)(p + 16);
        }
        #pragma unroll
        for (int ni = 0; ni < 2; ++ni) {
            const bf16_t* p = &sB[cur][(wn * 32 + ni * 16 + rsel) * LDT + hsel * 8];
            bfrag[ni].q[0] = *(const uint4*)p;
            bfrag[ni].q[1] = *(const uint4*)(p + 16);
        }
        #pragma unroll
        for (int mi = 0; mi < 4; ++mi)
            #pragma unroll
            for (int ni = 0; ni < 2; ++ni)
                acc[mi][ni] = wmma_bf16(afrag[mi].v, bfrag[ni].v, acc[mi][ni]);

        wait_async0();      // next tile landed in LDS
        __syncthreads();    // all lanes done reading cur, writes visible
    }

    // epilogue
    float* dst = (mode == 0) ? out_plain
               : (which == 0) ? dst0 : (which == 1) ? dst1 : dst2;
    bf16_t* bfdst = (mode == 1 && which == 1) ? kb : nullptr;

    #pragma unroll
    for (int mi = 0; mi < 4; ++mi) {
        #pragma unroll
        for (int ni = 0; ni < 2; ++ni) {
            int n = n0 + wn * 32 + ni * 16 + rsel;
            #pragma unroll
            for (int r = 0; r < 8; ++r) {
                int m = m0 + wm * 64 + mi * 16 + hsel * 8 + r;
                float val = acc[mi][ni][r];
                if (mode == 0) {
                    dst[(size_t)m * HD + n] = val;
                } else {
                    int b = m >> 12, l = m & (LL - 1);
                    int h = n >> 6,  d = n & (DD - 1);
                    size_t o = (((size_t)(b * NH + h)) * LL + l) * DD + d;
                    dst[o] = val;
                    if (bfdst) bfdst[o] = (bf16_t)val;
                }
            }
        }
    }
}

// ---------------------------------------------------------------------------
// 3) Sampled-score metric: M[b,h,l] = max_s(q.k_idx) - mean_s(q.k_idx)
// ---------------------------------------------------------------------------
__global__ __launch_bounds__(256) void metric_kernel(const float* __restrict__ qf,
                                                     const float* __restrict__ kf,
                                                     const int* __restrict__ idx,
                                                     float* __restrict__ metric) {
    __shared__ __align__(16) float sq[4][DD];
    __shared__ float sdot[4][SS];
    int t  = threadIdx.x;
    int rl = t >> 6;                       // 4 rows per block
    int s  = t & 63;
    size_t row = (size_t)blockIdx.x * 4 + rl;     // 0..BH*L-1
    int l = (int)(row & (LL - 1));
    sq[rl][s] = qf[row * DD + s];
    __syncthreads();
    if (s < SS) {
        int ksrc = idx[l * SS + s];
        size_t bh = row >> 12;
        const float4* k4 = (const float4*)(kf + (bh * LL + ksrc) * DD);
        const float4* q4 = (const float4*)sq[rl];
        float acc = 0.f;
        #pragma unroll
        for (int i = 0; i < 16; ++i) {
            float4 a = q4[i], b = k4[i];
            acc += a.x * b.x + a.y * b.y + a.z * b.z + a.w * b.w;
        }
        sdot[rl][s] = acc;
    }
    __syncthreads();
    if (s == 0) {
        float mx = -INFINITY, sum = 0.f;
        for (int i = 0; i < SS; ++i) { float v = sdot[rl][i]; mx = fmaxf(mx, v); sum += v; }
        metric[row] = mx - sum * (1.0f / SS);
    }
}

// ---------------------------------------------------------------------------
// 4) top-45 per (b,h): iterative argmax, ties -> lower index (matches top_k)
// ---------------------------------------------------------------------------
__global__ __launch_bounds__(256) void topk_kernel(const float* __restrict__ metric,
                                                   int* __restrict__ topidx) {
    __shared__ float vals[LL];
    __shared__ float rv[256];
    __shared__ int   ri[256];
    int bh = blockIdx.x, t = threadIdx.x;
    for (int i = t; i < LL; i += 256) vals[i] = metric[(size_t)bh * LL + i];
    __syncthreads();
    for (int it = 0; it < SS; ++it) {
        float bv = -INFINITY; int bi = 0x7fffffff;
        #pragma unroll
        for (int j = 0; j < 16; ++j) {
            int i = t * 16 + j;
            float v = vals[i];
            if (v > bv || (v == bv && i < bi)) { bv = v; bi = i; }
        }
        rv[t] = bv; ri[t] = bi;
        __syncthreads();
        for (int off = 128; off > 0; off >>= 1) {
            if (t < off) {
                float v2 = rv[t + off]; int i2 = ri[t + off];
                if (v2 > rv[t] || (v2 == rv[t] && i2 < ri[t])) { rv[t] = v2; ri[t] = i2; }
            }
            __syncthreads();
        }
        if (t == 0) { topidx[bh * UP + it] = ri[0]; vals[ri[0]] = -INFINITY; }
        __syncthreads();
    }
    if (t < UP - SS) topidx[bh * UP + SS + t] = 0;   // deterministic padding
}

// ---------------------------------------------------------------------------
// 5a) Gather Qr (48x64, rows >=45 zero) as bf16
// ---------------------------------------------------------------------------
__global__ __launch_bounds__(256) void gather_qr_kernel(const float* __restrict__ qf,
                                                        const int* __restrict__ topidx,
                                                        bf16_t* __restrict__ qr) {
    int bh = blockIdx.x, t = threadIdx.x;
    for (int i = t; i < UP * DD; i += 256) {
        int u = i >> 6, d = i & 63;
        float v = 0.f;
        if (u < SS) {
            int l = topidx[bh * UP + u];
            v = qf[((size_t)bh * LL + l) * DD + d];
        }
        qr[(size_t)bh * UP * DD + i] = (bf16_t)v;
    }
}

// ---------------------------------------------------------------------------
// 5b) scores(48 x 4096) = Qr @ k^T * scale  (WMMA, K = 64, async staging)
// ---------------------------------------------------------------------------
__global__ __launch_bounds__(256) void scores_kernel(const bf16_t* __restrict__ qr,
                                                     const bf16_t* __restrict__ kb,
                                                     float* __restrict__ scores) {
    __shared__ bf16_t sQ[UP * 72];
    __shared__ bf16_t sK[128 * 72];
    int bh = blockIdx.x;
    int lt = blockIdx.y;                   // l tile of 128
    int t = threadIdx.x, lane = t & 31, wid = t >> 5;

    // stage Qr: 48*64 bf16 = 384 x 16B chunks
    {
        int r = t >> 3, c = t & 7;
        async_copy_b128(qr + (size_t)bh * UP * DD + r * DD + c * 8, &sQ[r * 72 + c * 8]);
        if (t < 128) {
            int i = t + 256, r2 = i >> 3, c2 = i & 7;
            async_copy_b128(qr + (size_t)bh * UP * DD + r2 * DD + c2 * 8, &sQ[r2 * 72 + c2 * 8]);
        }
    }
    // stage kb chunk: 128*64 bf16 = 1024 x 16B chunks
    const bf16_t* kbase = kb + ((size_t)bh * LL + lt * 128) * DD;
    #pragma unroll
    for (int j = 0; j < 4; ++j) {
        int i = t + j * 256;
        int r = i >> 3, c = i & 7;
        async_copy_b128(kbase + (size_t)r * DD + c * 8, &sK[r * 72 + c * 8]);
    }
    wait_async0();
    __syncthreads();

    const int rsel = lane & 15, hsel = lane >> 4;
    v8f acc[3] = {};
    #pragma unroll
    for (int ks = 0; ks < 2; ++ks) {
        FragBF16 bfr;
        const bf16_t* pb = &sK[(wid * 16 + rsel) * 72 + ks * 32 + hsel * 8];
        bfr.q[0] = *(const uint4*)pb; bfr.q[1] = *(const uint4*)(pb + 16);
        #pragma unroll
        for (int mi = 0; mi < 3; ++mi) {
            FragBF16 af;
            const bf16_t* pa = &sQ[(mi * 16 + rsel) * 72 + ks * 32 + hsel * 8];
            af.q[0] = *(const uint4*)pa; af.q[1] = *(const uint4*)(pa + 16);
            acc[mi] = wmma_bf16(af.v, bfr.v, acc[mi]);
        }
    }
    const float scale = 0.125f;            // 1/sqrt(64)
    #pragma unroll
    for (int mi = 0; mi < 3; ++mi) {
        int n = lt * 128 + wid * 16 + rsel;
        #pragma unroll
        for (int r = 0; r < 8; ++r) {
            int u = mi * 16 + hsel * 8 + r;
            scores[((size_t)bh * UP + u) * LL + n] = acc[mi][r] * scale;
        }
    }
}

// ---------------------------------------------------------------------------
// 5c) row softmax over L; bf16 output; rows >= 45 zeroed
// ---------------------------------------------------------------------------
__global__ __launch_bounds__(256) void softmax_kernel(const float* __restrict__ scores,
                                                      bf16_t* __restrict__ attnb) {
    int bh = blockIdx.x, u = blockIdx.y, t = threadIdx.x;
    bf16_t* orow = attnb + ((size_t)bh * UP + u) * LL;
    if (u >= SS) {
        for (int i = t; i < LL; i += 256) orow[i] = (bf16_t)0.f;
        return;
    }
    const float* row = scores + ((size_t)bh * UP + u) * LL;
    __shared__ float red[256];
    float mx = -INFINITY;
    for (int i = t; i < LL; i += 256) mx = fmaxf(mx, row[i]);
    red[t] = mx; __syncthreads();
    for (int off = 128; off > 0; off >>= 1) {
        if (t < off) red[t] = fmaxf(red[t], red[t + off]);
        __syncthreads();
    }
    mx = red[0]; __syncthreads();
    float sum = 0.f;
    for (int i = t; i < LL; i += 256) sum += __expf(row[i] - mx);
    red[t] = sum; __syncthreads();
    for (int off = 128; off > 0; off >>= 1) {
        if (t < off) red[t] += red[t + off];
        __syncthreads();
    }
    float inv = 1.0f / red[0];
    for (int i = t; i < LL; i += 256) orow[i] = (bf16_t)(__expf(row[i] - mx) * inv);
}

// ---------------------------------------------------------------------------
// 5d) ctx(48 x 64) = attn @ v   (WMMA, K loop over L, v transposed into LDS)
// ---------------------------------------------------------------------------
__global__ __launch_bounds__(128) void ctx_kernel(const bf16_t* __restrict__ attnb,
                                                  const float* __restrict__ vf,
                                                  float* __restrict__ ctx) {
    __shared__ bf16_t sVT[DD * 40];        // [d][l-chunk of 32]
    int bh = blockIdx.x, t = threadIdx.x, lane = t & 31, wid = t >> 5;
    const int rsel = lane & 15, hsel = lane >> 4;
    v8f acc[3] = {};
    const float*  vbase = vf + (size_t)bh * LL * DD;
    const bf16_t* abase = attnb + (size_t)bh * UP * LL;

    for (int l0 = 0; l0 < LL; l0 += 32) {
        __syncthreads();
        // stage v chunk transposed (f32 -> bf16), 16 elems/thread, batched loads
        float tmp[16];
        #pragma unroll
        for (int j = 0; j < 16; ++j) {
            int i = t + j * 128;
            int l = i & 31, d = i >> 5;
            tmp[j] = vbase[(size_t)(l0 + l) * DD + d];
        }
        #pragma unroll
        for (int j = 0; j < 16; ++j) {
            int i = t + j * 128;
            int l = i & 31, d = i >> 5;
            sVT[d * 40 + l] = (bf16_t)tmp[j];
        }
        __syncthreads();
        FragBF16 bfr;
        const bf16_t* pb = &sVT[(wid * 16 + rsel) * 40 + hsel * 8];
        bfr.q[0] = *(const uint4*)pb; bfr.q[1] = *(const uint4*)(pb + 16);
        #pragma unroll
        for (int mi = 0; mi < 3; ++mi) {
            FragBF16 af;
            const bf16_t* pa = abase + (size_t)(mi * 16 + rsel) * LL + l0 + hsel * 8;
            af.q[0] = *(const uint4*)pa; af.q[1] = *(const uint4*)(pa + 16);
            acc[mi] = wmma_bf16(af.v, bfr.v, acc[mi]);
        }
    }
    #pragma unroll
    for (int mi = 0; mi < 3; ++mi) {
        int d = wid * 16 + rsel;
        #pragma unroll
        for (int r = 0; r < 8; ++r) {
            int u = mi * 16 + hsel * 8 + r;
            ctx[((size_t)bh * UP + u) * DD + d] = acc[mi][r];
        }
    }
}

// ---------------------------------------------------------------------------
// 6) v mean over L, broadcast context, scatter selected rows
// ---------------------------------------------------------------------------
__global__ __launch_bounds__(64) void vmean_kernel(const float* __restrict__ vf,
                                                   float* __restrict__ vmean) {
    int bh = blockIdx.x, d = threadIdx.x;
    const float* vb = vf + (size_t)bh * LL * DD;
    float s = 0.f;
    for (int l = 0; l < LL; ++l) s += vb[(size_t)l * DD + d];
    vmean[bh * DD + d] = s * (1.0f / LL);
}

__global__ __launch_bounds__(256) void build_ctx_kernel(const float* __restrict__ vmean,
                                                        bf16_t* __restrict__ ctxfull) {
    size_t i = (size_t)blockIdx.x * 256 + threadIdx.x;   // over MTOT*HD
    int n = (int)(i & 511);
    int m = (int)(i >> 9);
    int b = m >> 12, h = n >> 6, d = n & 63;
    ctxfull[i] = (bf16_t)vmean[(b * NH + h) * DD + d];
}

__global__ __launch_bounds__(256) void scatter_ctx_kernel(const float* __restrict__ ctx,
                                                          const int* __restrict__ topidx,
                                                          bf16_t* __restrict__ ctxfull) {
    int bh = blockIdx.x, t = threadIdx.x;
    int b = bh >> 3, h = bh & 7;
    for (int i = t; i < SS * DD; i += 256) {
        int u = i >> 6, d = i & 63;
        int l = topidx[bh * UP + u];
        ctxfull[((size_t)(b * LL + l)) * HD + h * DD + d] =
            (bf16_t)ctx[((size_t)bh * UP + u) * DD + d];
    }
}

// ---------------------------------------------------------------------------
// launcher
// ---------------------------------------------------------------------------
extern "C" void kernel_launch(void* const* d_in, const int* in_sizes, int n_in,
                              void* d_out, int out_size, void* d_ws, size_t ws_size,
                              hipStream_t stream) {
    const float* x   = (const float*)d_in[0];
    const float* Wq  = (const float*)d_in[1];
    const float* Wk  = (const float*)d_in[2];
    const float* Wv  = (const float*)d_in[3];
    const float* Wo  = (const float*)d_in[4];
    const int*   idx = (const int*)d_in[5];
    float* out = (float*)d_out;

    char* ws = (char*)d_ws;
    size_t off = 0;
    auto alloc = [&](size_t bytes) -> void* {
        off = (off + 255) & ~(size_t)255;
        void* p = ws + off;
        off += bytes;
        return p;
    };

    bf16_t* xb      = (bf16_t*)alloc((size_t)MTOT * HD * 2);
    bf16_t* WT      = (bf16_t*)alloc((size_t)4 * HD * HD * 2);
    float*  qf      = (float*) alloc((size_t)MTOT * HD * 4);
    float*  kf      = (float*) alloc((size_t)MTOT * HD * 4);
    float*  vf      = (float*) alloc((size_t)MTOT * HD * 4);
    bf16_t* kb      = (bf16_t*)alloc((size_t)MTOT * HD * 2);
    float*  metric  = (float*) alloc((size_t)BH * LL * 4);
    int*    topidx  = (int*)   alloc((size_t)BH * UP * 4);
    bf16_t* qr      = (bf16_t*)alloc((size_t)BH * UP * DD * 2);
    float*  scores  = (float*) alloc((size_t)BH * UP * LL * 4);
    bf16_t* attnb   = (bf16_t*)alloc((size_t)BH * UP * LL * 2);
    float*  ctx     = (float*) alloc((size_t)BH * UP * DD * 4);
    float*  vmean   = (float*) alloc((size_t)BH * DD * 4);
    bf16_t* ctxfull = (bf16_t*)alloc((size_t)MTOT * HD * 2);

    bf16_t* WTq = WT;
    bf16_t* WTk = WT + (size_t)1 * HD * HD;
    bf16_t* WTv = WT + (size_t)2 * HD * HD;
    bf16_t* WTo = WT + (size_t)3 * HD * HD;

    // 1) conversions
    convert_x_kernel<<<(MTOT * HD) / 256, 256, 0, stream>>>(x, xb);
    convert_w_kernel<<<dim3((HD * HD) / 256, 4), 256, 0, stream>>>(Wq, Wk, Wv, Wo, WT);

    // 2) q,k,v projections (mode 1, z selects weight)
    gemm_bf16_kernel<<<dim3(MTOT / 128, HD / 128, 3), 256, 0, stream>>>(
        xb, WTq, WTk, WTv, qf, kf, vf, kb, nullptr, 1);

    // 3) sampled-score metric
    metric_kernel<<<(BH * LL) / 4, 256, 0, stream>>>(qf, kf, idx, metric);

    // 4) top-45 per (b,h)
    topk_kernel<<<BH, 256, 0, stream>>>(metric, topidx);

    // 5) sparse attention
    gather_qr_kernel<<<BH, 256, 0, stream>>>(qf, topidx, qr);
    scores_kernel<<<dim3(BH, LL / 128), 256, 0, stream>>>(qr, kb, scores);
    softmax_kernel<<<dim3(BH, UP), 256, 0, stream>>>(scores, attnb);
    ctx_kernel<<<BH, 128, 0, stream>>>(attnb, vf, ctx);

    // 6) context assembly + output projection
    vmean_kernel<<<BH, 64, 0, stream>>>(vf, vmean);
    build_ctx_kernel<<<(MTOT * HD) / 256, 256, 0, stream>>>(vmean, ctxfull);
    scatter_ctx_kernel<<<BH, 256, 0, stream>>>(ctx, topidx, ctxfull);
    gemm_bf16_kernel<<<dim3(MTOT / 128, HD / 128, 1), 256, 0, stream>>>(
        ctxfull, WTo, nullptr, nullptr, nullptr, nullptr, nullptr, nullptr, out, 0);
}
